// BayesianKalmanNet_11819749998689
// MI455X (gfx1250) — matrix operand order repaired and checked by reading, not compiled, not run on hardware
//
#include <hip/hip_runtime.h>
#include <hip/hip_bf16.h>
#include <stdint.h>

// ---------------- CDNA5 WMMA types ----------------
typedef __attribute__((ext_vector_type(16))) __bf16 v16bf;
typedef __attribute__((ext_vector_type(8)))  __bf16 v8bf;
typedef __attribute__((ext_vector_type(8)))  float  v8f;

// ---------------- problem constants ----------------
#define BATCH   256
#define TSTEPS  512
#define SDIM    8
#define ODIM    4
#define HID     800          // HIDDEN
#define G3      2400         // 3*HID
#define INDIM   12           // SDIM+ODIM
#define OUTD    32           // SDIM*ODIM
#define BT      16           // batch tile per workgroup (WMMA M)
#define NWG     (BATCH / BT) // 16 workgroups
#define NTHREADS 256         // 8 wave32s
#define NTILES  (HID / 16)   // 50 column tiles of h
#define MAXTPW  7            // ceil(50/8) tiles per wave

// ---------------- helpers ----------------
__device__ __forceinline__ uint32_t hash32(uint32_t x) {
  x ^= x >> 16; x *= 0x7feb352dU;
  x ^= x >> 15; x *= 0x846ca68bU;
  x ^= x >> 16;
  return x;
}
// deterministic stateless dropout (keep=0.8, scale 1.25)
__device__ __forceinline__ float dropoutf(float v, uint32_t key, uint32_t idx) {
  uint32_t h = hash32(idx * 0x9e3779b9U + key * 0x85ebca6bU + 0x165667b1U);
  float u = (float)(h >> 8) * (1.0f / 16777216.0f);
  return (u < 0.8f) ? v * 1.25f : 0.0f;
}
__device__ __forceinline__ float sigm(float x) { return 1.0f / (1.0f + __expf(-x)); }
__device__ __forceinline__ float tanh_fast(float x) {
  x = fminf(fmaxf(x, -15.0f), 15.0f);
  float e = __expf(2.0f * x);
  return (e - 1.0f) / (e + 1.0f);
}

// fp32 -> bf16 weight conversion (one-shot pre-pass into d_ws)
__global__ void cvt_bf16_kernel(const float* __restrict__ src,
                                __bf16* __restrict__ dst, int n) {
  int i = blockIdx.x * blockDim.x + threadIdx.x;
  if (i < n) dst[i] = (__bf16)src[i];
}

// ---------------- persistent recurrent kernel ----------------
// grid = 16 WGs (one per 16-row batch tile), block = 256 (8 wave32s)
__global__ __launch_bounds__(NTHREADS, 1)
void kalman_gru_kernel(const float* __restrict__ y_seq,   // [B,T,O]
                       const float* __restrict__ W_in,    // [H,12] f32
                       const float* __restrict__ b_in,    // [H]
                       const float* __restrict__ b_ih,    // [3H]
                       const float* __restrict__ b_hh,    // [3H]
                       const float* __restrict__ b_out,   // [32]
                       const __bf16* __restrict__ WihB,   // [3H,H] bf16
                       const __bf16* __restrict__ WhhB,   // [3H,H] bf16
                       const __bf16* __restrict__ WoutB,  // [32,H] bf16
                       float* __restrict__ out)           // [B,T,S]
{
  extern __shared__ char smem[];
  float*  hF   = (float*)smem;                       // [BT*H] f32   51200 B
  __bf16* hB   = (__bf16*)(hF + BT * HID);           // [BT*H] bf16  25600 B
  __bf16* actB = hB + BT * HID;                      // [BT*H] bf16  25600 B
  float*  xprev= (float*)(actB + BT * HID);          // [BT*S]
  float*  dxs  = xprev + BT * SDIM;                  // [BT*S]
  float*  xpri = dxs   + BT * SDIM;                  // [BT*S]
  float*  innov= xpri  + BT * SDIM;                  // [BT*O]
  float*  nnin = innov + BT * ODIM;                  // [BT*12]
  float*  kvec = nnin  + BT * INDIM;                 // [BT*32]

  const int tid  = threadIdx.x;
  const int wave = tid >> 5;
  const int lane = tid & 31;
  const int m    = lane & 15;   // WMMA row (A) / column (B,D) within tile
  const int hi   = lane >> 4;   // lane-half select
  const int b0   = blockIdx.x * BT;

  // zero-init recurrent state
  for (int i = tid; i < BT * HID; i += NTHREADS) { hF[i] = 0.0f; hB[i] = (__bf16)0.0f; }
  for (int i = tid; i < BT * SDIM; i += NTHREADS) { xprev[i] = 0.0f; dxs[i] = 0.0f; }
  __syncthreads();

  for (int t = 0; t < TSTEPS; ++t) {
    // ---- phase 1: dynamics, innovation, l2-normalized NN input (1 thread/row)
    if (tid < BT) {
      const int r = tid;
      float nd = 0.0f, ni = 0.0f;
      #pragma unroll
      for (int s = 0; s < SDIM; ++s) {
        float x = xprev[r * SDIM + s];
        xpri[r * SDIM + s] = x + 0.1f * __sinf(x);
      }
      #pragma unroll
      for (int o = 0; o < ODIM; ++o) {
        float v = y_seq[(size_t)(b0 + r) * TSTEPS * ODIM + (size_t)t * ODIM + o]
                  - xpri[r * SDIM + o];
        innov[r * ODIM + o] = v;
        ni += v * v;
      }
      #pragma unroll
      for (int s = 0; s < SDIM; ++s) { float d = dxs[r * SDIM + s]; nd += d * d; }
      float rnd = 1.0f / fmaxf(sqrtf(nd), 1e-12f);
      float rni = 1.0f / fmaxf(sqrtf(ni), 1e-12f);
      #pragma unroll
      for (int s = 0; s < SDIM; ++s) nnin[r * INDIM + s] = dxs[r * SDIM + s] * rnd;
      #pragma unroll
      for (int o = 0; o < ODIM; ++o) nnin[r * INDIM + SDIM + o] = innov[r * ODIM + o] * rni;
    }
    __syncthreads();

    // ---- phase 2: act = relu(dropout(nnin @ W_in^T + b_in)) -> bf16 LDS
    for (int i = tid; i < BT * HID; i += NTHREADS) {
      const int r = i / HID, j = i - r * HID;
      float acc = b_in[j];
      #pragma unroll
      for (int kk = 0; kk < INDIM; ++kk) acc += nnin[r * INDIM + kk] * W_in[j * INDIM + kk];
      acc = dropoutf(acc, 2u * (uint32_t)t, (uint32_t)((b0 + r) * HID + j));
      actB[i] = (__bf16)fmaxf(acc, 0.0f);
    }
    __syncthreads();

    // ---- phase 3: fused GRU gate GEMMs via WMMA bf16; h_new kept in VGPRs
    v8f hnewT[MAXTPW];
    #pragma unroll
    for (int it = 0; it < MAXTPW; ++it) {
      const int tt = wave + 8 * it;
      if (tt < NTILES) {
        const int c = tt * 16;                 // h-column tile base
        v8f accR = {}, accZ = {}, accNX = {}, accNH = {};
        for (int k0 = 0; k0 < HID; k0 += 32) {
          // A fragments (16x32 bf16): lane m holds 16 contiguous K starting at 16*hi
          union { v16bf v; v8bf h2[2]; } ua, uh;
          ua.h2[0] = *(const v8bf*)(actB + m * HID + k0 + 16 * hi);
          ua.h2[1] = *(const v8bf*)(actB + m * HID + k0 + 16 * hi + 8);
          uh.h2[0] = *(const v8bf*)(hB   + m * HID + k0 + 16 * hi);
          uh.h2[1] = *(const v8bf*)(hB   + m * HID + k0 + 16 * hi + 8);
          // B fragments (32x16 bf16): lane m = output column = weight row c+m
          const size_t bo = (size_t)(c + m) * HID + (size_t)(k0 + 16 * hi);
          v16bf bR  = *(const v16bf*)(WihB + bo);
          v16bf bZ  = *(const v16bf*)(WihB + (size_t)HID * HID + bo);
          v16bf bNX = *(const v16bf*)(WihB + (size_t)2 * HID * HID + bo);
          v16bf cR  = *(const v16bf*)(WhhB + bo);
          v16bf cZ  = *(const v16bf*)(WhhB + (size_t)HID * HID + bo);
          v16bf cNH = *(const v16bf*)(WhhB + (size_t)2 * HID * HID + bo);
          accR  = __builtin_amdgcn_wmma_f32_16x16x32_bf16(false, ua.v, false, bR,  (short)0, accR,  false, false);
          accZ  = __builtin_amdgcn_wmma_f32_16x16x32_bf16(false, ua.v, false, bZ,  (short)0, accZ,  false, false);
          accNX = __builtin_amdgcn_wmma_f32_16x16x32_bf16(false, ua.v, false, bNX, (short)0, accNX, false, false);
          accR  = __builtin_amdgcn_wmma_f32_16x16x32_bf16(false, uh.v, false, cR,  (short)0, accR,  false, false);
          accZ  = __builtin_amdgcn_wmma_f32_16x16x32_bf16(false, uh.v, false, cZ,  (short)0, accZ,  false, false);
          accNH = __builtin_amdgcn_wmma_f32_16x16x32_bf16(false, uh.v, false, cNH, (short)0, accNH, false, false);
        }
        // gates: r,z sigmoid; n = tanh(gx_n + r*gh_n); h' = (1-z)*n + z*h
        const int nCol = c + m;
        const float br  = b_ih[nCol]        + b_hh[nCol];
        const float bz  = b_ih[HID + nCol]  + b_hh[HID + nCol];
        const float bnx = b_ih[2 * HID + nCol];
        const float bnh = b_hh[2 * HID + nCol];
        v8f hn;
        #pragma unroll
        for (int q = 0; q < 8; ++q) {
          const int M = q + 8 * hi;
          float rg = sigm(accR[q] + br);
          float zg = sigm(accZ[q] + bz);
          float ng = tanh_fast(accNX[q] + bnx + rg * (accNH[q] + bnh));
          hn[q] = (1.0f - zg) * ng + zg * hF[M * HID + nCol];
        }
        hnewT[it] = hn;
      }
    }
    __syncthreads();   // everyone done reading old h / act

    // write back h_new (f32 + bf16 mirror)
    #pragma unroll
    for (int it = 0; it < MAXTPW; ++it) {
      const int tt = wave + 8 * it;
      if (tt < NTILES) {
        const int nCol = tt * 16 + m;
        v8f hn = hnewT[it];
        #pragma unroll
        for (int q = 0; q < 8; ++q) {
          const int M = q + 8 * hi;
          hF[M * HID + nCol] = hn[q];
          hB[M * HID + nCol] = (__bf16)hn[q];
        }
      }
    }
    __syncthreads();

    // ---- phase 4: K_vec = h_new @ W_out^T + b_out  (2 WMMA column tiles)
    if (wave < 2) {
      const int c = wave * 16;
      v8f acc = {};
      for (int k0 = 0; k0 < HID; k0 += 32) {
        union { v16bf v; v8bf h2[2]; } uh;
        uh.h2[0] = *(const v8bf*)(hB + m * HID + k0 + 16 * hi);
        uh.h2[1] = *(const v8bf*)(hB + m * HID + k0 + 16 * hi + 8);
        v16bf bW = *(const v16bf*)(WoutB + (size_t)(c + m) * HID + (size_t)(k0 + 16 * hi));
        acc = __builtin_amdgcn_wmma_f32_16x16x32_bf16(false, uh.v, false, bW, (short)0, acc, false, false);
      }
      const int nCol = c + m;
      const float bb = b_out[nCol];
      #pragma unroll
      for (int q = 0; q < 8; ++q) kvec[(q + 8 * hi) * OUTD + nCol] = acc[q] + bb;
    }
    __syncthreads();

    // ---- phase 5: dropout(K_vec), correction einsum, state update, output
    if (tid < BT * SDIM) {
      const int r = tid >> 3, s = tid & 7;
      float corr = 0.0f;
      #pragma unroll
      for (int o = 0; o < ODIM; ++o) {
        const int j = s * ODIM + o;
        float kv = dropoutf(kvec[r * OUTD + j], 2u * (uint32_t)t + 1u,
                            (uint32_t)((b0 + r) * OUTD + j));
        corr += kv * innov[r * ODIM + o];
      }
      const float xp = xpri[r * SDIM + s] + corr;
      out[(size_t)(b0 + r) * TSTEPS * SDIM + (size_t)t * SDIM + s] = xp;
      xprev[r * SDIM + s] = xp;
      dxs[r * SDIM + s]   = corr;
    }
    __syncthreads();
  }
}

// ---------------- host launch ----------------
extern "C" void kernel_launch(void* const* d_in, const int* in_sizes, int n_in,
                              void* d_out, int out_size, void* d_ws, size_t ws_size,
                              hipStream_t stream) {
  const float* y_seq = (const float*)d_in[0];
  const float* W_in  = (const float*)d_in[1];
  const float* b_in  = (const float*)d_in[2];
  const float* W_ih  = (const float*)d_in[3];
  const float* W_hh  = (const float*)d_in[4];
  const float* b_ih  = (const float*)d_in[5];
  const float* b_hh  = (const float*)d_in[6];
  const float* W_out = (const float*)d_in[7];
  const float* b_out = (const float*)d_in[8];
  float* out = (float*)d_out;

  const size_t nIH = (size_t)G3 * HID;          // 1,920,000
  const size_t nOUT = (size_t)OUTD * HID;       // 25,600
  __bf16* WihB  = (__bf16*)d_ws;
  __bf16* WhhB  = (__bf16*)((char*)d_ws + nIH * 2);
  __bf16* WoutB = (__bf16*)((char*)d_ws + 2 * nIH * 2);

  cvt_bf16_kernel<<<(int)((nIH + 255) / 256), 256, 0, stream>>>(W_ih, WihB, (int)nIH);
  cvt_bf16_kernel<<<(int)((nIH + 255) / 256), 256, 0, stream>>>(W_hh, WhhB, (int)nIH);
  cvt_bf16_kernel<<<(int)((nOUT + 255) / 256), 256, 0, stream>>>(W_out, WoutB, (int)nOUT);

  const size_t ldsBytes =
      (size_t)BT * HID * 4 +          // hF
      (size_t)BT * HID * 2 +          // hB
      (size_t)BT * HID * 2 +          // actB
      (size_t)BT * SDIM * 4 * 3 +     // xprev, dxs, xpri
      (size_t)BT * ODIM * 4 +         // innov
      (size_t)BT * INDIM * 4 +        // nnin
      (size_t)BT * OUTD * 4;          // kvec

  kalman_gru_kernel<<<NWG, NTHREADS, ldsBytes, stream>>>(
      y_seq, W_in, b_in, b_ih, b_hh, b_out, WihB, WhhB, WoutB, out);
}
